// MambaEncoder_16776142258507
// MI455X (gfx1250) — matrix operand order repaired and checked by reading, not compiled, not run on hardware
//
#include <hip/hip_runtime.h>
#include <math.h>

// ---------------- model constants ----------------
#define VOCAB   32000
#define SEQ     2048
#define BATCH   4
#define D_MODEL 768
#define D_STATE 16
#define D_CONV  4
#define D_INNER 1536
#define DT_RANK 48
#define N_LAYERS 4
#define NTOK    (BATCH*SEQ)   // 8192 rows for all GEMMs

// ---------------- vector types -------------------
typedef __attribute__((ext_vector_type(16))) __bf16 v16bf;
typedef __attribute__((ext_vector_type(8)))  __bf16 v8bf;
typedef __attribute__((ext_vector_type(4)))  __bf16 v4bf;
typedef __attribute__((ext_vector_type(8)))  float  v8f;
typedef int v4i_vs __attribute__((vector_size(4 * sizeof(int))));

// ---------------- CDNA5 async LDS copy path -------------------
#if __has_builtin(__builtin_amdgcn_global_load_async_to_lds_b128) && \
    __has_builtin(__builtin_amdgcn_s_wait_asynccnt)
#define USE_ASYNC_LDS 1
#else
#define USE_ASYNC_LDS 0
#endif

#if USE_ASYNC_LDS
__device__ __forceinline__ void async_cp16(const __bf16* g, __bf16* l)
{
    // 16B per lane: GLOBAL_LOAD_ASYNC_TO_LDS_B128 (tracked by ASYNCcnt)
    __builtin_amdgcn_global_load_async_to_lds_b128(
        (__attribute__((address_space(1))) v4i_vs*)(void*)(const_cast<__bf16*>(g)),
        (__attribute__((address_space(3))) v4i_vs*)(void*)l,
        0, 0);
}
#endif

// =================================================================
// f32 -> bf16 conversion (weights pre-converted once per launch;
// bf16 weight set ~68MB -> resident in MI455X 192MB L2)
// =================================================================
__global__ __launch_bounds__(256) void cvt_bf16_kernel(
    const float* __restrict__ in, __bf16* __restrict__ out, size_t n)
{
    size_t i = ((size_t)blockIdx.x * 256 + threadIdx.x) * 4;
    if (i + 4 <= n) {
        float4 f = *(const float4*)(in + i);
        v4bf v;
        v[0] = (__bf16)f.x; v[1] = (__bf16)f.y;
        v[2] = (__bf16)f.z; v[3] = (__bf16)f.w;
        *(v4bf*)(out + i) = v;
    } else {
        for (; i < n; ++i) out[i] = (__bf16)in[i];
    }
}

// =================================================================
// Embedding: h[b,l,:] = tok_emb[x[b,l],:] + pos_emb[l,:]
// =================================================================
__global__ __launch_bounds__(256) void embed_kernel(
    const int* __restrict__ x, const float* __restrict__ tok,
    const float* __restrict__ pos, float* __restrict__ h)
{
    size_t i = (size_t)blockIdx.x * 256 + threadIdx.x;
    if (i >= (size_t)NTOK * D_MODEL) return;
    int c = (int)(i % D_MODEL);
    size_t row = i / D_MODEL;          // b*SEQ + l
    int l = (int)(row % SEQ);
    int t = x[row];
    h[i] = tok[(size_t)t * D_MODEL + c] + pos[(size_t)l * D_MODEL + c];
}

// =================================================================
// LayerNorm -> bf16 output (consumed only by WMMA GEMMs)
// one 256-thread block (8 wave32) per token row
// =================================================================
__global__ __launch_bounds__(256) void layernorm_kernel(
    const float* __restrict__ in, const float* __restrict__ g,
    const float* __restrict__ b, __bf16* __restrict__ out)
{
    __shared__ float rs[8], rs2[8], stat[2];
    const size_t row = blockIdx.x;
    const float* xr = in + row * D_MODEL;
    float s = 0.f, s2 = 0.f;
    for (int c = threadIdx.x; c < D_MODEL; c += 256) {
        float v = xr[c]; s += v; s2 += v * v;
    }
    for (int off = 16; off > 0; off >>= 1) {
        s  += __shfl_down(s,  off, 32);
        s2 += __shfl_down(s2, off, 32);
    }
    int lane = threadIdx.x & 31, wave = threadIdx.x >> 5;
    if (lane == 0) { rs[wave] = s; rs2[wave] = s2; }
    __syncthreads();
    if (threadIdx.x == 0) {
        float S = 0.f, S2 = 0.f;
        for (int w = 0; w < 8; ++w) { S += rs[w]; S2 += rs2[w]; }
        float mu  = S / D_MODEL;
        float var = S2 / D_MODEL - mu * mu;
        stat[0] = mu; stat[1] = rsqrtf(var + 1e-5f);
    }
    __syncthreads();
    float mu = stat[0], rstd = stat[1];
    __bf16* orow = out + row * D_MODEL;
    for (int c = threadIdx.x; c < D_MODEL; c += 256)
        orow[c] = (__bf16)((xr[c] - mu) * rstd * g[c] + b[c]);
}

// =================================================================
// WMMA GEMM: out(M,N) = A(M,K) * W(N,K)^T, bf16 in / f32 accumulate.
// Block tile 128x128, TK=32, 8 waves each owning 32x64 ->
// 8 x v_wmma_f32_16x16x32_bf16 per K-step.
// Full tiles staged via GLOBAL_LOAD_ASYNC_TO_LDS_B128 (ASYNCcnt),
// partial tiles via guarded sync loads with zero fill.
// Epilogues: 0 store, 2 +bias relu (bf16 out when OB), 3 +bias softplus,
//            4 residual add, 5 residual add +bias
// =================================================================
#define TM 128
#define TN 128
#define TK 32
#define LDSS 40   // padded LDS row stride (elements); 80B keeps 16B align

template<int EPI, int OB>
__global__ __launch_bounds__(256) void gemm_wmma_bf16(
    const __bf16* __restrict__ A, int lda,
    const __bf16* __restrict__ W, int ldw,
    const float* __restrict__ bias,
    void* __restrict__ outp, int ldout,
    int M, int N, int K)
{
    __shared__ __bf16 shA[TM * LDSS];
    __shared__ __bf16 shB[TN * LDSS];

    const int tid  = threadIdx.x;
    const int lane = tid & 31;
    const int wave = tid >> 5;
    const int wm   = wave & 3;   // 4 wave rows -> 128 M
    const int wn   = wave >> 2;  // 2 wave cols x 64 -> 128 N
    const int m0   = blockIdx.y * TM;
    const int n0   = blockIdx.x * TN;
    const int lrow = lane & 15;
    const int lhi  = lane >> 4;

    const v8f vz = {0.f,0.f,0.f,0.f,0.f,0.f,0.f,0.f};
    v8f acc[2][4];
#pragma unroll
    for (int i = 0; i < 2; ++i)
#pragma unroll
        for (int j = 0; j < 4; ++j) acc[i][j] = vz;

    const bool aInb = (m0 + TM) <= M;
    const bool bInb = (n0 + TN) <= N;

    for (int kt = 0; kt < K; kt += TK) {
        const bool fullk = (kt + TK) <= K;
        const bool asyncA = (USE_ASYNC_LDS != 0) && fullk && aInb;
        const bool asyncB = (USE_ASYNC_LDS != 0) && fullk && bInb;

        // ---- stage A tile (128x32 bf16): 512 x 16B chunks ----
        for (int i = tid; i < (TM * TK) / 8; i += 256) {
            int r = i >> 2;
            int c = (i & 3) * 8;
            int gm = m0 + r;
#if USE_ASYNC_LDS
            if (asyncA) {
                async_cp16(A + (size_t)gm * lda + kt + c, &shA[r * LDSS + c]);
            } else
#endif
            {
                v8bf v;
                if (gm < M && fullk) {
                    v = *(const v8bf*)(A + (size_t)gm * lda + kt + c);
                } else {
#pragma unroll
                    for (int j = 0; j < 8; ++j) {
                        int gk = kt + c + j;
                        v[j] = (gm < M && gk < K) ? A[(size_t)gm * lda + gk]
                                                  : (__bf16)0.f;
                    }
                }
                *(v8bf*)&shA[r * LDSS + c] = v;
            }
        }
        // ---- stage W tile (128x32 bf16) ----
        for (int i = tid; i < (TN * TK) / 8; i += 256) {
            int r = i >> 2;
            int c = (i & 3) * 8;
            int gn = n0 + r;
#if USE_ASYNC_LDS
            if (asyncB) {
                async_cp16(W + (size_t)gn * ldw + kt + c, &shB[r * LDSS + c]);
            } else
#endif
            {
                v8bf v;
                if (gn < N && fullk) {
                    v = *(const v8bf*)(W + (size_t)gn * ldw + kt + c);
                } else {
#pragma unroll
                    for (int j = 0; j < 8; ++j) {
                        int gk = kt + c + j;
                        v[j] = (gn < N && gk < K) ? W[(size_t)gn * ldw + gk]
                                                  : (__bf16)0.f;
                    }
                }
                *(v8bf*)&shB[r * LDSS + c] = v;
            }
        }
#if USE_ASYNC_LDS
        if (asyncA || asyncB) __builtin_amdgcn_s_wait_asynccnt(0);
#endif
        __syncthreads();

        // ---- fragments per ISA VGPR layouts ----
        // A (16x32): lanes 0-15 K{0..7,16..23}, lanes 16-31 K{8..15,24..31}
        const int ka = lhi ? 8 : 0;
        v16bf af[2];
#pragma unroll
        for (int fm = 0; fm < 2; ++fm) {
            int r = wm * 32 + fm * 16 + lrow;
            v8bf lo = *(const v8bf*)&shA[r * LDSS + ka];
            v8bf hi = *(const v8bf*)&shA[r * LDSS + 16 + ka];
            af[fm] = __builtin_shufflevector(lo, hi,
                0,1,2,3,4,5,6,7,8,9,10,11,12,13,14,15);
        }
        // B (32x16): lanes 0-15 K{0..7,16..23}, lanes 16-31 K{8..15,24..31}
        const int kb = lhi ? 8 : 0;
        v16bf bfr[4];
#pragma unroll
        for (int fn = 0; fn < 4; ++fn) {
            int r = wn * 64 + fn * 16 + lrow;
            v8bf lo = *(const v8bf*)&shB[r * LDSS + kb];
            v8bf hi = *(const v8bf*)&shB[r * LDSS + 16 + kb];
            bfr[fn] = __builtin_shufflevector(lo, hi,
                0,1,2,3,4,5,6,7,8,9,10,11,12,13,14,15);
        }
#pragma unroll
        for (int fm = 0; fm < 2; ++fm)
#pragma unroll
            for (int fn = 0; fn < 4; ++fn)
                acc[fm][fn] = __builtin_amdgcn_wmma_f32_16x16x32_bf16(
                    false, af[fm], false, bfr[fn],
                    (short)0, acc[fm][fn], false, false);
        __syncthreads();
    }

    // ---- epilogue: C/D layout VGPR r -> M = r (+8 hi half), N = lane&15
    float*  outf = (float*)outp;
    __bf16* outh = (__bf16*)outp;
#pragma unroll
    for (int fm = 0; fm < 2; ++fm) {
#pragma unroll
        for (int fn = 0; fn < 4; ++fn) {
            int n = n0 + wn * 64 + fn * 16 + lrow;
            if (n >= N) continue;
            float bv = 0.f;
            if (EPI == 2 || EPI == 3 || EPI == 5) bv = bias[n];
            int mbase = m0 + wm * 32 + fm * 16 + lhi * 8;
#pragma unroll
            for (int r = 0; r < 8; ++r) {
                int m = mbase + r;
                if (m >= M) continue;
                float v = acc[fm][fn][r];
                size_t o = (size_t)m * ldout + n;
                if (EPI == 0) {
                    if (OB) outh[o] = (__bf16)v; else outf[o] = v;
                } else if (EPI == 2) {
                    v += bv; v = v > 0.f ? v : 0.f;
                    if (OB) outh[o] = (__bf16)v; else outf[o] = v;
                } else if (EPI == 3) {
                    v += bv;
                    outf[o] = (v > 20.f) ? v : log1pf(expf(v));
                } else if (EPI == 4) {
                    outf[o] += v;
                } else {                    // EPI == 5
                    outf[o] += v + bv;
                }
            }
        }
    }
}

// =================================================================
// Causal depthwise conv (width 4) + SiLU; writes f32 (scan) and
// bf16 (GEMM-A) copies.
// =================================================================
__global__ __launch_bounds__(256) void conv_silu_kernel(
    const float* __restrict__ xz, const float* __restrict__ cw,
    const float* __restrict__ cb, float* __restrict__ xc,
    __bf16* __restrict__ xcb)
{
    size_t i = (size_t)blockIdx.x * 256 + threadIdx.x;
    if (i >= (size_t)NTOK * D_INNER) return;
    int d = (int)(i % D_INNER);
    size_t row = i / D_INNER;          // b*SEQ + l
    int l = (int)(row % SEQ);
    size_t b = row / SEQ;
    float acc = cb[d];
#pragma unroll
    for (int k = 0; k < D_CONV; ++k) {
        int ls = l - (D_CONV - 1) + k;
        if (ls >= 0)
            acc += xz[(b * SEQ + ls) * (2 * D_INNER) + d] * cw[d * D_CONV + k];
    }
    float s = acc / (1.f + expf(-acc));   // silu
    xc[i]  = s;
    xcb[i] = (__bf16)s;
}

// =================================================================
// Selective scan: one thread per (b,d); 16-state recurrence in regs,
// fused y = (scan + x*D) * silu(z); bf16 output (feeds out_proj GEMM)
// =================================================================
__global__ __launch_bounds__(256) void scan_kernel(
    const float* __restrict__ xc, const float* __restrict__ dt,
    const float* __restrict__ proj, const float* __restrict__ xz,
    const float* __restrict__ A_log, const float* __restrict__ Dp,
    __bf16* __restrict__ y)
{
    int t = blockIdx.x * 256 + threadIdx.x;
    if (t >= BATCH * D_INNER) return;
    int b = t / D_INNER, d = t % D_INNER;

    float A[D_STATE], h[D_STATE];
#pragma unroll
    for (int s = 0; s < D_STATE; ++s) {
        A[s] = -expf(A_log[d * D_STATE + s]);
        h[s] = 0.f;
    }
    float Dv = Dp[d];
    const int PW = DT_RANK + 2 * D_STATE;   // 80
    for (int l = 0; l < SEQ; ++l) {
        size_t row = (size_t)b * SEQ + l;
        float dtv = dt[row * D_INNER + d];
        float xv  = xc[row * D_INNER + d];
        const float* pr = proj + row * PW;
        float dx = dtv * xv;
        float yv = 0.f;
#pragma unroll
        for (int s = 0; s < D_STATE; ++s) {
            float dA = expf(dtv * A[s]);
            h[s] = dA * h[s] + dx * pr[DT_RANK + s];
            yv  += h[s] * pr[DT_RANK + D_STATE + s];
        }
        yv += xv * Dv;
        float zv = xz[row * (2 * D_INNER) + D_INNER + d];
        yv *= zv / (1.f + expf(-zv));
        y[row * D_INNER + d] = (__bf16)yv;
    }
}

// =================================================================
// host-side dispatch
// =================================================================
static void launch_gemm(int epi, const __bf16* A, int lda, const __bf16* W, int ldw,
                        const float* bias, void* out, int ldout,
                        int M, int N, int K, hipStream_t s)
{
    dim3 grid((N + TN - 1) / TN, (M + TM - 1) / TM);
    dim3 block(256);
    switch (epi) {
    case 0: gemm_wmma_bf16<0,0><<<grid, block, 0, s>>>(A, lda, W, ldw, bias, out, ldout, M, N, K); break;
    case 2: gemm_wmma_bf16<2,1><<<grid, block, 0, s>>>(A, lda, W, ldw, bias, out, ldout, M, N, K); break;
    case 3: gemm_wmma_bf16<3,0><<<grid, block, 0, s>>>(A, lda, W, ldw, bias, out, ldout, M, N, K); break;
    case 4: gemm_wmma_bf16<4,0><<<grid, block, 0, s>>>(A, lda, W, ldw, bias, out, ldout, M, N, K); break;
    default: gemm_wmma_bf16<5,0><<<grid, block, 0, s>>>(A, lda, W, ldw, bias, out, ldout, M, N, K); break;
    }
}

static void launch_cvt(const float* in, __bf16* out, size_t n, hipStream_t s)
{
    cvt_bf16_kernel<<<(unsigned)((n / 4 + 255) / 256), 256, 0, s>>>(in, out, n);
}

extern "C" void kernel_launch(void* const* d_in, const int* in_sizes, int n_in,
                              void* d_out, int out_size, void* d_ws, size_t ws_size,
                              hipStream_t stream)
{
    (void)in_sizes; (void)n_in; (void)out_size; (void)ws_size;
    const int*   x        = (const int*)  d_in[0];
    const float* tok_emb  = (const float*)d_in[1];
    const float* pos_emb  = (const float*)d_in[2];
    const float* ln1_g    = (const float*)d_in[3];
    const float* ln1_b    = (const float*)d_in[4];
    const float* in_w     = (const float*)d_in[5];
    const float* conv_w   = (const float*)d_in[6];
    const float* conv_b   = (const float*)d_in[7];
    const float* xproj_w  = (const float*)d_in[8];
    const float* dtproj_w = (const float*)d_in[9];
    const float* dtproj_b = (const float*)d_in[10];
    const float* A_log    = (const float*)d_in[11];
    const float* Dp       = (const float*)d_in[12];
    const float* out_w    = (const float*)d_in[13];
    const float* ln2_g    = (const float*)d_in[14];
    const float* ln2_b    = (const float*)d_in[15];
    const float* w1       = (const float*)d_in[16];
    const float* b1       = (const float*)d_in[17];
    const float* w2       = (const float*)d_in[18];
    const float* b2       = (const float*)d_in[19];

    float* h = (float*)d_out;                   // residual stream (NTOK x 768)

    // ---------------- workspace layout ----------------
    // f32 buffers (scan-precision path)
    float* xz   = (float*)d_ws;                         // NTOK*3072
    float* xc   = xz   + (size_t)NTOK * (2 * D_INNER);  // NTOK*1536
    float* dt   = xc   + (size_t)NTOK * D_INNER;        // NTOK*1536
    float* proj = dt   + (size_t)NTOK * D_INNER;        // NTOK*80
    // bf16 buffers (WMMA GEMM operands)
    __bf16* lnb   = (__bf16*)(proj + (size_t)NTOK * 80);       // NTOK*768
    __bf16* xcb   = lnb   + (size_t)NTOK * D_MODEL;            // NTOK*1536
    __bf16* yb    = xcb   + (size_t)NTOK * D_INNER;            // NTOK*1536
    __bf16* projb = yb    + (size_t)NTOK * D_INNER;            // NTOK*80
    __bf16* wb    = projb + (size_t)NTOK * 80;                 // bf16 weights
    __bf16* ffb   = xcb;   // FFN mid (NTOK*3072) aliases xcb+yb (mamba done)

    const size_t SZ_INW  = (size_t)N_LAYERS * (2 * D_INNER) * D_MODEL;
    const size_t SZ_XPJ  = (size_t)N_LAYERS * (DT_RANK + 2 * D_STATE) * D_INNER;
    const size_t SZ_DTW  = (size_t)N_LAYERS * D_INNER * DT_RANK;
    const size_t SZ_OUTW = (size_t)N_LAYERS * D_MODEL * D_INNER;
    const size_t SZ_W1   = (size_t)N_LAYERS * (4 * D_MODEL) * D_MODEL;
    const size_t SZ_W2   = (size_t)N_LAYERS * D_MODEL * (4 * D_MODEL);
    __bf16* in_wb  = wb;
    __bf16* xprojb = in_wb  + SZ_INW;
    __bf16* dtwb   = xprojb + SZ_XPJ;
    __bf16* outwb  = dtwb   + SZ_DTW;
    __bf16* w1b    = outwb  + SZ_OUTW;
    __bf16* w2b    = w1b    + SZ_W1;

    // ---- pre-convert all weights to bf16 (once per launch) ----
    launch_cvt(in_w,     in_wb,  SZ_INW,  stream);
    launch_cvt(xproj_w,  xprojb, SZ_XPJ,  stream);
    launch_cvt(dtproj_w, dtwb,   SZ_DTW,  stream);
    launch_cvt(out_w,    outwb,  SZ_OUTW, stream);
    launch_cvt(w1,       w1b,    SZ_W1,   stream);
    launch_cvt(w2,       w2b,    SZ_W2,   stream);

    // ---- embedding ----
    {
        size_t tot = (size_t)NTOK * D_MODEL;
        embed_kernel<<<(unsigned)((tot + 255) / 256), 256, 0, stream>>>(x, tok_emb, pos_emb, h);
    }

    for (int L = 0; L < N_LAYERS; ++L) {
        const __bf16* in_w_l  = in_wb  + (size_t)L * (2 * D_INNER) * D_MODEL;
        const __bf16* xproj_l = xprojb + (size_t)L * (DT_RANK + 2 * D_STATE) * D_INNER;
        const __bf16* dtw_l   = dtwb   + (size_t)L * D_INNER * DT_RANK;
        const __bf16* out_w_l = outwb  + (size_t)L * D_MODEL * D_INNER;
        const __bf16* w1_l    = w1b    + (size_t)L * (4 * D_MODEL) * D_MODEL;
        const __bf16* w2_l    = w2b    + (size_t)L * D_MODEL * (4 * D_MODEL);
        const float* conv_w_l = conv_w + (size_t)L * D_INNER * D_CONV;
        const float* conv_b_l = conv_b + (size_t)L * D_INNER;
        const float* dtb_l    = dtproj_b + (size_t)L * D_INNER;
        const float* Alog_l   = A_log  + (size_t)L * D_INNER * D_STATE;
        const float* Dp_l     = Dp     + (size_t)L * D_INNER;
        const float* b1_l     = b1     + (size_t)L * (4 * D_MODEL);
        const float* b2_l     = b2     + (size_t)L * D_MODEL;

        // --- mamba branch ---
        layernorm_kernel<<<NTOK, 256, 0, stream>>>(h, ln1_g + L * D_MODEL, ln1_b + L * D_MODEL, lnb);
        // xz = ln1(h) @ in_w^T  (8192 x 3072, K=768)
        launch_gemm(0, lnb, D_MODEL, in_w_l, D_MODEL, nullptr, xz, 2 * D_INNER,
                    NTOK, 2 * D_INNER, D_MODEL, stream);
        // causal conv + silu on x half
        {
            size_t tot = (size_t)NTOK * D_INNER;
            conv_silu_kernel<<<(unsigned)((tot + 255) / 256), 256, 0, stream>>>(
                xz, conv_w_l, conv_b_l, xc, xcb);
        }
        // proj = xc @ xproj_w^T  (8192 x 80, K=1536)
        launch_gemm(0, xcb, D_INNER, xproj_l, D_INNER, nullptr, proj, DT_RANK + 2 * D_STATE,
                    NTOK, DT_RANK + 2 * D_STATE, D_INNER, stream);
        // bf16 copy of proj for dtproj GEMM A
        launch_cvt(proj, projb, (size_t)NTOK * (DT_RANK + 2 * D_STATE), stream);
        // dt = softplus(dtr @ dtproj_w^T + b)  (8192 x 1536, K=48, lda=80)
        launch_gemm(3, projb, DT_RANK + 2 * D_STATE, dtw_l, DT_RANK, dtb_l, dt, D_INNER,
                    NTOK, D_INNER, DT_RANK, stream);
        // selective scan fused with * silu(z)
        scan_kernel<<<(BATCH * D_INNER + 255) / 256, 256, 0, stream>>>(
            xc, dt, proj, xz, Alog_l, Dp_l, yb);
        // h += y @ out_w^T  (8192 x 768, K=1536)
        launch_gemm(4, yb, D_INNER, out_w_l, D_INNER, nullptr, h, D_MODEL,
                    NTOK, D_MODEL, D_INNER, stream);

        // --- FFN branch ---
        layernorm_kernel<<<NTOK, 256, 0, stream>>>(h, ln2_g + L * D_MODEL, ln2_b + L * D_MODEL, lnb);
        // ff = relu(ln2(h) @ w1^T + b1)  (8192 x 3072, K=768) -> bf16
        launch_gemm(2, lnb, D_MODEL, w1_l, D_MODEL, b1_l, ffb, 4 * D_MODEL,
                    NTOK, 4 * D_MODEL, D_MODEL, stream);
        // h += ff @ w2^T + b2  (8192 x 768, K=3072)
        launch_gemm(5, ffb, 4 * D_MODEL, w2_l, 4 * D_MODEL, b2_l, h, D_MODEL,
                    NTOK, D_MODEL, 4 * D_MODEL, stream);
    }
}
